// MLLABlock_86371792322779
// MI455X (gfx1250) — matrix operand order repaired and checked by reading, not compiled, not run on hardware
//
#include <hip/hip_runtime.h>
#include <hip/hip_bf16.h>
#include <math.h>

// ---------------------------------------------------------------------------
// MLLA block for MI455X (gfx1250, wave32). All GEMMs use
// v_wmma_f32_16x16x32_bf16 (bf16 operands, f32 accumulate).
// Dense GEMMs stage the per-block weight slab in LDS (shared by 8 waves).
// ---------------------------------------------------------------------------

typedef __bf16 bf16_t;
typedef __attribute__((ext_vector_type(16))) __bf16 v16bf;
typedef __attribute__((ext_vector_type(8)))  __bf16 v8bf;
typedef __attribute__((ext_vector_type(8)))  float  v8f;

#define CDIM 128
#define LDIM 16384   // H*W = 128*128

__device__ __forceinline__ float gelu_f(float x) {
    return 0.5f * x * (1.0f + erff(x * 0.70710678118654752f));
}

// Pack WMMA A operand: elements 0..7 = K[cb..cb+7], 8..15 = K[cb+16..cb+23]
// (ISA 7.12.2, 16-bit A 16x32: VGPR0-3 hold first 8 K, VGPR4-7 hold K+16 group)
__device__ __forceinline__ v16bf ldA16(const bf16_t* p) {
    v8bf lo = *reinterpret_cast<const v8bf*>(p);
    v8bf hi = *reinterpret_cast<const v8bf*>(p + 16);
    v16bf r;
#pragma unroll
    for (int i = 0; i < 8; ++i) { r[i] = lo[i]; r[i + 8] = hi[i]; }
    return r;
}

// Pack WMMA B operand: 16 contiguous halfs (K = kbase..kbase+15 for column n)
__device__ __forceinline__ v16bf ldB16(const bf16_t* p) {
    v8bf lo = *reinterpret_cast<const v8bf*>(p);
    v8bf hi = *reinterpret_cast<const v8bf*>(p + 8);
    v16bf r;
#pragma unroll
    for (int i = 0; i < 8; ++i) { r[i] = lo[i]; r[i + 8] = hi[i]; }
    return r;
}

// ---------------------------------------------------------------------------
// Generic bf16 GEMM:  out[M,N] = A[M,K] @ W[N,K]^T  with fused epilogues.
// Block = 256 threads = 8 waves; wave -> one 16-row x 128-col tile.
// The 128xK weight slab for this block is staged through LDS in 128-k chunks
// (32KB), shared by all 8 waves -> B operands become immediate-offset
// ds_load_b128 and weight L2 traffic drops 8x.
// grid = (M/128, N/128).
// EPI: 0 store f32 | 1 GELU->f32 | 2 ELU+1 -> split q/k bf16 (N=256)
//      3 *z *act_res -> bf16 (per-batch W) | 4 +resid -> f32
//      5 +bias,GELU -> bf16 | 6 +bias +resid -> f32
// ---------------------------------------------------------------------------
template <int EPI>
__global__ __launch_bounds__(256) void gemm_kernel(
    const bf16_t* __restrict__ A, const bf16_t* __restrict__ W,
    float* __restrict__ outF, bf16_t* __restrict__ outB, bf16_t* __restrict__ outB2,
    const float* __restrict__ bias, const float* __restrict__ resid,
    const float* __restrict__ zmul, const float* __restrict__ emul,
    int N, int K, int wBatchStride, int Lb)
{
    __shared__ bf16_t Wl[128 * 128];   // 32KB weight slab chunk

    const int lane = threadIdx.x & 31;
    const int wv   = threadIdx.x >> 5;
    const int lr   = lane & 15;      // row/col within 16
    const int lh   = lane >> 4;      // half-wave select
    const int mTile = (blockIdx.x * 8 + wv) * 16;
    const int nBase = blockIdx.y * 128;

    // batch is uniform per block (128 rows never straddle a batch: Lb % 128 == 0)
    const bf16_t* Wb = W + (size_t)((blockIdx.x * 128) / Lb) * (size_t)wBatchStride;

    v8f zacc = {0.f, 0.f, 0.f, 0.f, 0.f, 0.f, 0.f, 0.f};
    v8f acc[8];
#pragma unroll
    for (int t = 0; t < 8; ++t) acc[t] = zacc;

    const bf16_t* aRow = A + (size_t)(mTile + lr) * K + lh * 8;

    for (int kc = 0; kc < K; kc += 128) {
        __syncthreads();
        // cooperative stage: rows nBase..nBase+127, k = kc..kc+127 (16B per copy)
        for (int i = threadIdx.x; i < 128 * 128 / 8; i += 256) {
            int row = i >> 4;              // i/16
            int col = (i & 15) * 8;
            *reinterpret_cast<v8bf*>(&Wl[row * 128 + col]) =
                *reinterpret_cast<const v8bf*>(&Wb[(size_t)(nBase + row) * K + kc + col]);
        }
        __syncthreads();

        for (int k0 = 0; k0 < 128; k0 += 32) {
            v16bf av = ldA16(aRow + kc + k0);
#pragma unroll
            for (int t = 0; t < 8; ++t) {
                v16bf bv = ldA16(&Wl[(t * 16 + lr) * 128 + k0 + lh * 8]);
                acc[t] = __builtin_amdgcn_wmma_f32_16x16x32_bf16(
                    false, av, false, bv, (short)0, acc[t], false, false);
            }
        }
    }

    // D layout (f32 16x16): lane col = lr, VGPR r -> row r + 8*lh
#pragma unroll
    for (int t = 0; t < 8; ++t) {
        const int col = nBase + t * 16 + lr;
#pragma unroll
        for (int r = 0; r < 8; ++r) {
            const int row = mTile + r + 8 * lh;
            const float v = acc[t][r];
            const size_t o = (size_t)row * N + col;
            if (EPI == 0) {
                outF[o] = v;
            } else if (EPI == 1) {
                outF[o] = gelu_f(v);
            } else if (EPI == 2) {
                float e = (v > 0.f) ? (v + 1.f) : expf(v);   // elu(x)+1
                size_t oq = (size_t)row * CDIM;
                if (col < CDIM) outB[oq + col] = (bf16_t)e;
                else            outB2[oq + col - CDIM] = (bf16_t)e;
            } else if (EPI == 3) {
                float s = v * zmul[row] * emul[(size_t)row * CDIM + col];
                outB[(size_t)row * CDIM + col] = (bf16_t)s;
            } else if (EPI == 4) {
                outF[o] = resid[o] + v;
            } else if (EPI == 5) {
                outB[o] = (bf16_t)gelu_f(v + bias[col]);
            } else { // 6
                outF[o] = resid[o] + v + bias[col];
            }
        }
    }
}

// ---------------------------------------------------------------------------
// kv^T[b][e][d] += sum_n xi[b,n,e] * k[b,n,d]   (split-K over L, f32 atomics)
// LDS-staged with transpose so A/B operand reads are contiguous ds_load_b128.
// ---------------------------------------------------------------------------
__global__ __launch_bounds__(256) void kv_kernel(
    const bf16_t* __restrict__ k, const bf16_t* __restrict__ xi,
    float* __restrict__ kvT, int L, int nchunk)
{
    __shared__ bf16_t kT[CDIM * 32];   // [d][n_local]
    __shared__ bf16_t xT[CDIM * 32];   // [e][n_local]

    const int lane = threadIdx.x & 31;
    const int wv   = threadIdx.x >> 5;
    const int lr   = lane & 15;
    const int lh   = lane >> 4;
    const int b    = blockIdx.x / nchunk;
    const int ch   = blockIdx.x % nchunk;
    const int span = L / nchunk;
    const int n0   = ch * span;
    const size_t base = (size_t)b * L;

    v8f zacc = {0.f, 0.f, 0.f, 0.f, 0.f, 0.f, 0.f, 0.f};
    v8f acc[8];
#pragma unroll
    for (int t = 0; t < 8; ++t) acc[t] = zacc;

    for (int s = 0; s < span; s += 32) {
        for (int i = threadIdx.x; i < 32 * CDIM; i += 256) {
            int nl = i >> 7, d = i & 127;
            size_t gi = (base + n0 + s + nl) * CDIM + d;
            kT[d * 32 + nl] = k[gi];
            xT[d * 32 + nl] = xi[gi];
        }
        __syncthreads();
        v16bf av = ldA16(&xT[(wv * 16 + lr) * 32 + lh * 8]);      // A = xi^T tile
#pragma unroll
        for (int t = 0; t < 8; ++t) {
            v16bf bv = ldA16(&kT[(t * 16 + lr) * 32 + lh * 8]);  // B = k tile
            acc[t] = __builtin_amdgcn_wmma_f32_16x16x32_bf16(
                false, av, false, bv, (short)0, acc[t], false, false);
        }
        __syncthreads();
    }

#pragma unroll
    for (int t = 0; t < 8; ++t) {
        const int d = t * 16 + lr;
#pragma unroll
        for (int r = 0; r < 8; ++r) {
            const int e = wv * 16 + r + 8 * lh;
            atomicAdd(&kvT[((size_t)b * CDIM + e) * CDIM + d], acc[t][r]);
        }
    }
}

// ---------------------------------------------------------------------------
// Depthwise 3x3 "SAME" conv on channel-last [B, H*W, C].
// MODE 0: outF = in + conv(in) + bias   (CPE)
// MODE 1: outB = bf16(GELU(conv(in) + bias))   (dwc branch)
// ---------------------------------------------------------------------------
template <int MODE>
__global__ __launch_bounds__(256) void dwconv_kernel(
    const float* __restrict__ in, const float* __restrict__ wconv,
    const float* __restrict__ bias, float* __restrict__ outF,
    bf16_t* __restrict__ outB)
{
    __shared__ float lw[CDIM * 9];
    __shared__ float lb[CDIM];
    for (int i = threadIdx.x; i < CDIM * 9; i += 256) lw[i] = wconv[i];
    if (threadIdx.x < CDIM) lb[threadIdx.x] = bias[threadIdx.x];
    __syncthreads();

    const size_t idx = (size_t)blockIdx.x * 256 + threadIdx.x;
    const int c = idx & 127;
    const int w = (idx >> 7) & 127;
    const int h = (idx >> 14) & 127;
    const int b = (int)(idx >> 21);

    float s = lb[c];
#pragma unroll
    for (int dh = 0; dh < 3; ++dh) {
        int hh = h + dh - 1;
        if ((unsigned)hh < 128u) {
#pragma unroll
            for (int dw = 0; dw < 3; ++dw) {
                int ww = w + dw - 1;
                if ((unsigned)ww < 128u)
                    s += in[(((size_t)b * LDIM) + hh * 128 + ww) * CDIM + c] *
                         lw[c * 9 + dh * 3 + dw];
            }
        }
    }
    if (MODE == 0) outF[idx] = in[idx] + s;
    else           outB[idx] = (bf16_t)gelu_f(s);
}

// LayerNorm over C=128, one wave per token -> bf16 out.
__global__ __launch_bounds__(256) void ln_kernel(
    const float* __restrict__ x, const float* __restrict__ g,
    const float* __restrict__ b, bf16_t* __restrict__ out)
{
    const int lane = threadIdx.x & 31;
    const int wv   = threadIdx.x >> 5;
    const size_t tok = (size_t)blockIdx.x * 8 + wv;
    const float4 v = *reinterpret_cast<const float4*>(x + tok * CDIM + lane * 4);
    float s  = v.x + v.y + v.z + v.w;
    float s2 = v.x * v.x + v.y * v.y + v.z * v.z + v.w * v.w;
#pragma unroll
    for (int m = 16; m; m >>= 1) { s += __shfl_xor(s, m, 32); s2 += __shfl_xor(s2, m, 32); }
    const float mean = s * (1.f / 128.f);
    const float var  = s2 * (1.f / 128.f) - mean * mean;
    const float rs   = rsqrtf(var + 1e-5f);
    const int c0 = lane * 4;
    const float vv[4] = {v.x, v.y, v.z, v.w};
#pragma unroll
    for (int i = 0; i < 4; ++i)
        out[tok * CDIM + c0 + i] = (bf16_t)((vv[i] - mean) * rs * g[c0 + i] + b[c0 + i]);
}

// ksum[b][d] = sum_n k[b,n,d]
__global__ __launch_bounds__(256) void ksum_kernel(
    const bf16_t* __restrict__ k, float* __restrict__ ksum, int L)
{
    __shared__ float red[256];
    const int b = blockIdx.x;
    const int d = threadIdx.x & 127, half = threadIdx.x >> 7;
    const size_t base = (size_t)b * L;
    float s = 0.f;
    for (int n = half; n < L; n += 2) s += (float)k[(base + n) * CDIM + d];
    red[threadIdx.x] = s;
    __syncthreads();
    if (threadIdx.x < 128) ksum[b * CDIM + d] = red[threadIdx.x] + red[threadIdx.x + 128];
}

// z[tok] = 1 / (q[tok,:] . ksum[b,:] + 1e-6)
__global__ __launch_bounds__(256) void z_kernel(
    const bf16_t* __restrict__ q, const float* __restrict__ ksum,
    float* __restrict__ z, int L)
{
    const int lane = threadIdx.x & 31;
    const int wv   = threadIdx.x >> 5;
    const size_t tok = (size_t)blockIdx.x * 8 + wv;
    const int b = (int)(tok / (size_t)L);
    const bf16_t* qp = q + tok * CDIM + lane * 4;
    const float*  kp = ksum + b * CDIM + lane * 4;
    float dot = 0.f;
#pragma unroll
    for (int i = 0; i < 4; ++i) dot += (float)qp[i] * kp[i];
#pragma unroll
    for (int m = 16; m; m >>= 1) dot += __shfl_xor(dot, m, 32);
    if (lane == 0) z[tok] = 1.f / (dot + 1e-6f);
}

// f32 -> bf16 conversion (weights, kv matrix)
__global__ __launch_bounds__(256) void cvt_kernel(
    const float* __restrict__ in, bf16_t* __restrict__ out, int n)
{
    int i = blockIdx.x * 256 + threadIdx.x;
    if (i < n) out[i] = (bf16_t)in[i];
}

// ---------------------------------------------------------------------------
extern "C" void kernel_launch(void* const* d_in, const int* in_sizes, int n_in,
                              void* d_out, int out_size, void* d_ws, size_t ws_size,
                              hipStream_t stream)
{
    const float* x      = (const float*)d_in[0];
    const float* cpe1_w = (const float*)d_in[3];
    const float* cpe1_b = (const float*)d_in[4];
    const float* cpe2_w = (const float*)d_in[5];
    const float* cpe2_b = (const float*)d_in[6];
    const float* n1g    = (const float*)d_in[7];
    const float* n1b    = (const float*)d_in[8];
    const float* n2g    = (const float*)d_in[9];
    const float* n2b    = (const float*)d_in[10];
    const float* wact   = (const float*)d_in[11];
    const float* win    = (const float*)d_in[12];
    const float* wout   = (const float*)d_in[13];
    const float* wqk    = (const float*)d_in[14];
    const float* dwc_w  = (const float*)d_in[15];
    const float* dwc_b  = (const float*)d_in[16];
    const float* fc1w   = (const float*)d_in[17];
    const float* fc1b   = (const float*)d_in[18];
    const float* fc2w   = (const float*)d_in[19];
    const float* fc2b   = (const float*)d_in[20];

    const int Bn = in_sizes[0] / (LDIM * CDIM);   // 8
    const int M  = Bn * LDIM;                     // 131072
    const size_t MB = 1ull << 20;
    char* ws = (char*)d_ws;

    // Scratch layout (aliased by lifetime; peak ~293 MB):
    float*  x1   = (float*)(ws + 0);          // 64MB  shortcut (CPE1 out)
    bf16_t* bufA = (bf16_t*)(ws + 64 * MB);   // 32MB  xn -> xi -> xn2
    float*  actb = (float*)(ws + 96 * MB);    // 64MB  act_res -> x3
    float*  xip  = (float*)(ws + 160 * MB);   // 64MB  xi_pre -> x2
    bf16_t* qb   = (bf16_t*)(ws + 224 * MB);  // 32MB  q
    bf16_t* kb   = (bf16_t*)(ws + 256 * MB);  // 32MB  k -> u
    bf16_t* hb   = (bf16_t*)(ws + 160 * MB);  // 128MB h (over dead xip/q/k)
    float*  ksumb= (float*)(ws + 288 * MB);
    float*  zb   = (float*)(ws + 289 * MB);
    float*  kvf  = (float*)(ws + 290 * MB);   // kv^T f32 partials
    bf16_t* kvb  = (bf16_t*)(ws + 291 * MB);  // kv^T bf16
    bf16_t* wa   = (bf16_t*)(ws + 292 * MB);
    bf16_t* wi   = wa  + 128 * 128;
    bf16_t* wo   = wi  + 128 * 128;
    bf16_t* wqkb = wo  + 128 * 128;
    bf16_t* w1   = wqkb + 256 * 128;
    bf16_t* w2   = w1  + 512 * 128;

    // Convert GEMM weights f32 -> bf16
    cvt_kernel<<<64, 256, 0, stream>>>(wact, wa, 128 * 128);
    cvt_kernel<<<64, 256, 0, stream>>>(win,  wi, 128 * 128);
    cvt_kernel<<<64, 256, 0, stream>>>(wout, wo, 128 * 128);
    cvt_kernel<<<128, 256, 0, stream>>>(wqk, wqkb, 256 * 128);
    cvt_kernel<<<256, 256, 0, stream>>>(fc1w, w1, 512 * 128);
    cvt_kernel<<<256, 256, 0, stream>>>(fc2w, w2, 128 * 512);

    hipMemsetAsync(kvf, 0, (size_t)Bn * CDIM * CDIM * sizeof(float), stream);

    const unsigned elemBlocks = (unsigned)((size_t)M * CDIM / 256);
    const dim3 g128(M / 128, 1), g256(M / 128, 2), g512(M / 128, 4);

    // 1. CPE1: x1 = x + dwconv(x)
    dwconv_kernel<0><<<elemBlocks, 256, 0, stream>>>(x, cpe1_w, cpe1_b, x1, nullptr);
    // 2. xn = LN1(x1)
    ln_kernel<<<M / 8, 256, 0, stream>>>(x1, n1g, n1b, bufA);
    // 3. act_res = GELU(xn @ wact^T)
    gemm_kernel<1><<<g128, 256, 0, stream>>>(bufA, wa, actb, nullptr, nullptr,
                                             nullptr, nullptr, nullptr, nullptr,
                                             128, 128, 0, LDIM);
    // 4. xi_pre = xn @ win^T
    gemm_kernel<0><<<g128, 256, 0, stream>>>(bufA, wi, xip, nullptr, nullptr,
                                             nullptr, nullptr, nullptr, nullptr,
                                             128, 128, 0, LDIM);
    // 5. xi = GELU(dwconv(xi_pre))  (bf16, overwrites xn)
    dwconv_kernel<1><<<elemBlocks, 256, 0, stream>>>(xip, dwc_w, dwc_b, nullptr, bufA);
    // 6. qk = xi @ wqk^T ; q,k = elu(qk)+1 (split)
    gemm_kernel<2><<<g256, 256, 0, stream>>>(bufA, wqkb, nullptr, qb, kb,
                                             nullptr, nullptr, nullptr, nullptr,
                                             256, 128, 0, LDIM);
    // 7. ksum
    ksum_kernel<<<Bn, 256, 0, stream>>>(kb, ksumb, LDIM);
    // 8. kv^T = xi^T-reduce (split-K, atomics) ; convert to bf16
    kv_kernel<<<Bn * 32, 256, 0, stream>>>(kb, bufA, kvf, LDIM, 32);
    cvt_kernel<<<(Bn * CDIM * CDIM + 255) / 256, 256, 0, stream>>>(kvf, kvb, Bn * CDIM * CDIM);
    // 9. z = 1/(q.ksum + 1e-6)
    z_kernel<<<M / 8, 256, 0, stream>>>(qb, ksumb, zb, LDIM);
    // 10. u = (q @ kv) * z * act_res   (per-batch kv weights; u into kb)
    gemm_kernel<3><<<g128, 256, 0, stream>>>(qb, kvb, nullptr, kb, nullptr,
                                             nullptr, nullptr, zb, actb,
                                             128, 128, CDIM * CDIM, LDIM);
    // 11. x2 = x1 + u @ wout^T   (into xip)
    gemm_kernel<4><<<g128, 256, 0, stream>>>(kb, wo, xip, nullptr, nullptr,
                                             nullptr, x1, nullptr, nullptr,
                                             128, 128, 0, LDIM);
    // 12. CPE2: x3 = x2 + dwconv(x2)  (into actb)
    dwconv_kernel<0><<<elemBlocks, 256, 0, stream>>>(xip, cpe2_w, cpe2_b, actb, nullptr);
    // 13. xn2 = LN2(x3)
    ln_kernel<<<M / 8, 256, 0, stream>>>(actb, n2g, n2b, bufA);
    // 14. h = GELU(xn2 @ fc1^T + b1)  (bf16)
    gemm_kernel<5><<<g512, 256, 0, stream>>>(bufA, w1, nullptr, hb, nullptr,
                                             fc1b, nullptr, nullptr, nullptr,
                                             512, 128, 0, LDIM);
    // 15. out = x3 + h @ fc2^T + b2
    gemm_kernel<6><<<g128, 256, 0, stream>>>(hb, w2, (float*)d_out, nullptr, nullptr,
                                             fc2b, actb, nullptr, nullptr,
                                             128, 512, 0, LDIM);
}
